// DiffusionNet_83932250898799
// MI455X (gfx1250) — matrix operand order repaired and checked by reading, not compiled, not run on hardware
//
#include <hip/hip_runtime.h>
#include <hip/hip_bf16.h>
#include <math.h>

typedef __attribute__((ext_vector_type(16))) __bf16 v16bf;
typedef __attribute__((ext_vector_type(8)))  __bf16 v8bf;
typedef __attribute__((ext_vector_type(8)))  float  v8f;

#define IN_DIM 288
#define H      1024
#define MT     64          // batch rows per workgroup (4 M-tiles of 16)
#define BLK    256         // 8 waves of 32
#define NW     8

// ---------------- preprocess: W[K][N] fp32 -> Wt[N][K] bf16 ----------------
__global__ void transpose_bf16(const float* __restrict__ W, __bf16* __restrict__ Wt,
                               int K, int N) {
  int idx = blockIdx.x * blockDim.x + threadIdx.x;
  if (idx >= K * N) return;
  int n = idx / K, k = idx % K;
  Wt[idx] = (__bf16)W[(size_t)k * N + n];
}

// W4 [1024][7] -> Wt4 [16][1024] bf16 (pad N 7->16 with zeros)
__global__ void transpose_pad_w4(const float* __restrict__ W4, __bf16* __restrict__ Wt) {
  int idx = blockIdx.x * blockDim.x + threadIdx.x;
  if (idx >= 16 * 1024) return;
  int n = idx / 1024, k = idx % 1024;
  float v = (n < 7) ? W4[k * 7 + n] : 0.f;
  Wt[idx] = (__bf16)v;
}

// ---------------- WMMA helpers ----------------
__device__ __forceinline__ v8f wmma_bf16(v16bf a, v16bf b, v8f c) {
  return __builtin_amdgcn_wmma_f32_16x16x32_bf16(false, a, false, b,
                                                 (short)0, c, false, false);
}

// Load one A fragment (16x32 bf16, contiguous k = 16*half + s) from row-major LDS (row stride H).
__device__ __forceinline__ v16bf load_a(const __bf16* in, int m0, int l16, int half,
                                        int k0) {
  const __bf16* arow = in + (m0 + l16) * H;
  v8bf alo = *(const v8bf*)(arow + k0 + 16 * half);
  v8bf ahi = *(const v8bf*)(arow + k0 + 16 * half + 8);
  v16bf a;
#pragma unroll
  for (int i = 0; i < 8; ++i) { a[i] = alo[i]; a[8 + i] = ahi[i]; }
  return a;
}

// ---- register-blocked layer: C[64, N=1024] = relu?(A[64,K] x W[K,1024] + b) ----
// Each wave: all 4 M-tiles x 2 N-tiles per sweep; 8 accumulators; B reused 4x, A 2x.
template<int K, bool RELU>
__device__ __forceinline__ void mlp_layer(const __bf16* in, __bf16* outb,
                                          const __bf16* __restrict__ wt,
                                          const float* __restrict__ bias,
                                          int wave, int lane) {
  const int half = lane >> 4;
  const int l16  = lane & 15;

  for (int g = wave; g < (H / 32); g += NW) {   // 32 groups of 2 N-tiles
    const int n0 = g * 32;
    const float bv0 = bias[n0 + l16];
    const float bv1 = bias[n0 + 16 + l16];

    v8f acc[4][2];
#pragma unroll
    for (int mt = 0; mt < 4; ++mt)
#pragma unroll
      for (int j = 0; j < 2; ++j)
#pragma unroll
        for (int i = 0; i < 8; ++i) acc[mt][j][i] = j ? bv1 : bv0;

    const __bf16* b0row = wt + (size_t)(n0 + l16) * K + 16 * half;
    const __bf16* b1row = wt + (size_t)(n0 + 16 + l16) * K + 16 * half;

#pragma unroll 2
    for (int k0 = 0; k0 < K; k0 += 32) {
      v16bf a0 = load_a(in, 0,  l16, half, k0);
      v16bf a1 = load_a(in, 16, l16, half, k0);
      v16bf a2 = load_a(in, 32, l16, half, k0);
      v16bf a3 = load_a(in, 48, l16, half, k0);
      v16bf b0 = *(const v16bf*)(b0row + k0);
      v16bf b1 = *(const v16bf*)(b1row + k0);
      if (k0 + 256 < K) {                       // prefetch next chunks of B into GL2/WGP$
        __builtin_prefetch(b0row + k0 + 256, 0, 1);
        __builtin_prefetch(b1row + k0 + 256, 0, 1);
      }
      acc[0][0] = wmma_bf16(a0, b0, acc[0][0]);
      acc[1][0] = wmma_bf16(a1, b0, acc[1][0]);
      acc[2][0] = wmma_bf16(a2, b0, acc[2][0]);
      acc[3][0] = wmma_bf16(a3, b0, acc[3][0]);
      acc[0][1] = wmma_bf16(a0, b1, acc[0][1]);
      acc[1][1] = wmma_bf16(a1, b1, acc[1][1]);
      acc[2][1] = wmma_bf16(a2, b1, acc[2][1]);
      acc[3][1] = wmma_bf16(a3, b1, acc[3][1]);
    }

#pragma unroll
    for (int mt = 0; mt < 4; ++mt)
#pragma unroll
      for (int j = 0; j < 2; ++j) {
        const int c = n0 + j * 16 + l16;
#pragma unroll
        for (int i = 0; i < 8; ++i) {
          float v = acc[mt][j][i];
          if (RELU) v = v > 0.f ? v : 0.f;
          outb[(mt * 16 + i + 8 * half) * H + c] = (__bf16)v;
        }
      }
  }
}

// single-tile GEMM for the tiny final layer (N=16)
template<int K>
__device__ __forceinline__ v8f tile_gemm(const __bf16* in, const __bf16* __restrict__ wt,
                                         int m0, int n0, int l16, int half, float bias) {
  v8f acc;
#pragma unroll
  for (int i = 0; i < 8; ++i) acc[i] = bias;
  const __bf16* brow = wt + (size_t)(n0 + l16) * K + 16 * half;
#pragma unroll 4
  for (int k0 = 0; k0 < K; k0 += 32) {
    v16bf a = load_a(in, m0, l16, half, k0);
    v16bf b = *(const v16bf*)(brow + k0);
    acc = wmma_bf16(a, b, acc);
  }
  return acc;
}

__launch_bounds__(BLK)
__global__ void fused_mlp(const float* __restrict__ x,
                          const __bf16* __restrict__ W1t, const float* __restrict__ b1,
                          const __bf16* __restrict__ W2t, const float* __restrict__ b2,
                          const __bf16* __restrict__ W3t, const float* __restrict__ b3,
                          const __bf16* __restrict__ W4t, const float* __restrict__ b4,
                          float* __restrict__ out, int batch) {
  __shared__ __align__(16) __bf16 bufA[MT * H];   // 128 KB
  __shared__ __align__(16) __bf16 bufB[MT * H];   // 128 KB
  __shared__ __align__(16) float  out7[MT * 16];  // 4 KB

  const int tid  = threadIdx.x;
  const int wave = tid >> 5;
  const int lane = tid & 31;
  const int half = lane >> 4;
  const int l16  = lane & 15;
  const int row0 = blockIdx.x * MT;

  // ---- stage x tile into LDS as bf16 (row stride H) ----
  for (int i = tid; i < MT * IN_DIM; i += BLK) {
    int r = i / IN_DIM, c = i % IN_DIM;
    bufA[r * H + c] = (__bf16)x[(size_t)(row0 + r) * IN_DIM + c];
  }
  __syncthreads();

  mlp_layer<IN_DIM, true>(bufA, bufB, W1t, b1, wave, lane);   // layer 1
  __syncthreads();
  mlp_layer<H, true>(bufB, bufA, W2t, b2, wave, lane);        // layer 2
  __syncthreads();
  mlp_layer<H, true>(bufA, bufB, W3t, b3, wave, lane);        // layer 3
  __syncthreads();

  // ---- layer 4: [MT,1024] x [1024,16(pad of 7)] -> out7 (f32, no relu) ----
  for (int t = wave; t < 4; t += NW) {          // waves 0..3; uniform per wave
    int m0 = t * 16;
    float bias = (l16 < 7) ? b4[l16] : 0.f;
    v8f acc = tile_gemm<H>(bufB, W4t, m0, 0, l16, half, bias);
#pragma unroll
    for (int i = 0; i < 8; ++i)
      out7[(m0 + i + 8 * half) * 16 + l16] = acc[i];
  }
  __syncthreads();

  // ---- per-row postprocess: rotations, eigvals, D = R E R^T ----
  if (tid < MT) {
    const float* o = &out7[tid * 16];
    float dd[6];
#pragma unroll
    for (int j = 0; j < 6; ++j) dd[j] = 1.f / (1.f + __expf(-o[j]));
    float S0v = tanhf(o[6]) + 1.f;

    const float PI = 3.14159265358979323846f;
    float ax = dd[0] * (2.f * PI), ay = dd[1] * PI, az = dd[2] * (2.f * PI);
    float e1 = dd[3] * 0.003f, e2 = e1 * dd[4], e3 = e2 * dd[5];

    float sx, cx, sy, cy, sz, cz;
    __sincosf(ax, &sx, &cx);
    __sincosf(ay, &sy, &cy);
    __sincosf(az, &sz, &cz);

    // R = Rz * Ry * Rx
    float R[3][3];
    R[0][0] = cz * cy; R[0][1] = cz * sy * sx - sz * cx; R[0][2] = cz * sy * cx + sz * sx;
    R[1][0] = sz * cy; R[1][1] = sz * sy * sx + cz * cx; R[1][2] = sz * sy * cx - cz * sx;
    R[2][0] = -sy;     R[2][1] = cy * sx;                R[2][2] = cy * cx;

    float ev[3] = { e1, e2, e3 };
    size_t row = (size_t)row0 + tid;
    size_t nb  = (size_t)batch;
    float* Ro = out;
    float* Eo = out + 9 * nb;
    float* Do = out + 18 * nb;
    float* So = out + 27 * nb;

#pragma unroll
    for (int i = 0; i < 3; ++i) {
#pragma unroll
      for (int j = 0; j < 3; ++j) {
        Ro[row * 9 + i * 3 + j] = R[i][j];
        Eo[row * 9 + i * 3 + j] = (i == j) ? ev[i] : 0.f;
        float dij = 0.f;
#pragma unroll
        for (int k = 0; k < 3; ++k) dij += R[i][k] * ev[k] * R[j][k];
        Do[row * 9 + i * 3 + j] = dij;
      }
    }
    So[row] = S0v;
  }
}

// ---------------- launch ----------------
extern "C" void kernel_launch(void* const* d_in, const int* in_sizes, int n_in,
                              void* d_out, int out_size, void* d_ws, size_t ws_size,
                              hipStream_t stream) {
  const float* x  = (const float*)d_in[0];
  const float* W1 = (const float*)d_in[1];
  const float* b1 = (const float*)d_in[2];
  const float* W2 = (const float*)d_in[3];
  const float* b2 = (const float*)d_in[4];
  const float* W3 = (const float*)d_in[5];
  const float* b3 = (const float*)d_in[6];
  const float* W4 = (const float*)d_in[7];
  const float* b4 = (const float*)d_in[8];
  float* out = (float*)d_out;
  int batch = in_sizes[0] / IN_DIM;

  char* ws = (char*)d_ws;
  const size_t szW1t = (size_t)1024 * 288 * 2;   // 32B-aligned
  const size_t szW2t = (size_t)1024 * 1024 * 2;  // 2 MB
  __bf16* W1t = (__bf16*)(ws);
  __bf16* W2t = (__bf16*)(ws + szW1t);
  __bf16* W3t = (__bf16*)(ws + szW1t + szW2t);
  __bf16* W4t = (__bf16*)(ws + szW1t + 2 * szW2t);

  transpose_bf16<<<(288 * 1024 + 255) / 256, 256, 0, stream>>>(W1, W1t, 288, 1024);
  transpose_bf16<<<(1024 * 1024 + 255) / 256, 256, 0, stream>>>(W2, W2t, 1024, 1024);
  transpose_bf16<<<(1024 * 1024 + 255) / 256, 256, 0, stream>>>(W3, W3t, 1024, 1024);
  transpose_pad_w4<<<(16 * 1024 + 255) / 256, 256, 0, stream>>>(W4, W4t);

  fused_mlp<<<batch / MT, BLK, 0, stream>>>(x, W1t, b1, W2t, b2, W3t, b3,
                                            W4t, b4, out, batch);
}
